// GPModuleLayer_85521388798635
// MI455X (gfx1250) — compile-verified, matrix operand order checked by reading, not run on hardware
//
#include <hip/hip_runtime.h>

// ---------------------------------------------------------------------------
// GPModuleLayer for MI455X (gfx1250, wave32, WMMA f32_16x16x32_f16,
// GLOBAL_LOAD_ASYNC_TO_LDS_B128 staging).
//
// Input index assumption (setup_inputs dict, tuples/dicts flattened in order):
//  0 X[16,32,200,64] 1 X_skip[16,64,200,1] 2 A_tilde(unused) 3 idx[200] 4 training
//  5 sci_w 6 sci_b 7..10 filt_w(k=2,3,6,7) 11..14 filt_b 15..18 gate_w 19..22 gate_b
//  23 skip_w[64,32,58] 24 skip_b
//  25..32 mp1{lin1_w,lin1_b,lin2_w,lin2_b,mlp1_w,mlp1_b,mlp2_w,mlp2_b}
//  33..40 mp2{...} 41 ln_w[32,200,58] 42 ln_b
// Output: Xo[16,32,200,58] then X_skip_out[16,64,200,1], flat f32.
// Workspace use ~104 MB (see offsets in kernel_launch).
// ---------------------------------------------------------------------------

typedef __attribute__((ext_vector_type(16))) _Float16 v16h;
typedef __attribute__((ext_vector_type(8)))  float    v8f;

#define NB   16
#define NC   32
#define NN   200
#define NTIME 64
#define TOUT 58
#define NPAD 224      // nodes padded to 14*16 (also K padded to 7*32)
#define NTL  14       // 16-wide n tiles
#define LS   228      // f32 LDS row stride: 912B = 57*16 (B128-aligned rows)
#define XHS  232      // f16 LDS row stride: 464B = 29*16 (B128-aligned rows)
#define ALPHA_F 0.05f

__device__ __forceinline__ int koffmap(int lane, int i) {
  // f16 A/B fragment: element i -> K offset (ISA 7.12.2, 16-bit A 16x32)
  return (lane < 16) ? (i < 8 ? i : i + 8) : (i < 8 ? i + 8 : i + 16);
}

__device__ __forceinline__ v8f wmma16(v16h a, v16h b, v8f c) {
  return __builtin_amdgcn_wmma_f32_16x16x32_f16(false, a, false, b, (short)0, c,
                                                false, false);
}

// Async memory->LDS copy, 16 bytes per lane (ASYNCcnt-tracked; ISA 10 op 98).
__device__ __forceinline__ void async_b128(unsigned lds_byte, const void* gaddr) {
  asm volatile("global_load_async_to_lds_b128 %0, %1, off"
               :: "v"(lds_byte), "v"(gaddr) : "memory");
}
__device__ __forceinline__ void wait_async0() {
#if __has_builtin(__builtin_amdgcn_s_wait_asynccnt)
  __builtin_amdgcn_s_wait_asynccnt(0);
#else
  asm volatile("s_wait_asynccnt 0x0" ::: "memory");
#endif
}
__device__ __forceinline__ unsigned lds_addr32(const void* p) {
  // generic LDS pointer -> 32-bit LDS byte address (ISA: LDS takes addr[31:0])
  return (unsigned)(unsigned long long)p;
}

// A[m=v][k=c] = x[c][v]  (x: f16 LDS, stride XHS)
__device__ __forceinline__ v16h fragA_xT(const _Float16* x, int vbase, int lane) {
  v16h a; int m = vbase + (lane & 15);
#pragma unroll
  for (int i = 0; i < 16; ++i) a[i] = x[koffmap(lane, i) * XHS + m];
  return a;
}
// B[k=c][n=w] = x[c][w]
__device__ __forceinline__ v16h fragB_x(const _Float16* x, int nbase, int lane) {
  v16h v; int n = nbase + (lane & 15);
#pragma unroll
  for (int i = 0; i < 16; ++i) v[i] = x[koffmap(lane, i) * XHS + n];
  return v;
}
// A[m=c][k=v] = h[c][v] * inv[v]  (h: f32 LDS stride LS; softmax scale folded in)
__device__ __forceinline__ v16h fragA_hscaled(const float* h, const float* inv,
                                              int mbase, int kbase, int lane) {
  v16h a; int m = mbase + (lane & 15);
#pragma unroll
  for (int i = 0; i < 16; ++i) {
    int k = kbase + koffmap(lane, i);
    a[i] = (_Float16)(h[m * LS + k] * inv[k]);
  }
  return a;
}
// B[k=v][n=w] = E[v][w]
__device__ __forceinline__ v16h fragB_E(const float* E, int kbase, int nbase, int lane) {
  v16h v; int n = nbase + (lane & 15);
#pragma unroll
  for (int i = 0; i < 16; ++i) v[i] = (_Float16)E[(kbase + koffmap(lane, i)) * LS + n];
  return v;
}
// B[k=v][n=w] = E[w][v]   (transposed propagation)
__device__ __forceinline__ v16h fragB_ET(const float* E, int kbase, int nbase, int lane) {
  v16h v; int n = nbase + (lane & 15);
#pragma unroll
  for (int i = 0; i < 16; ++i) v[i] = (_Float16)E[n * LS + kbase + koffmap(lane, i)];
  return v;
}
// A[m=o][k=c] = W[o][cbase+c]   (mlp weight block, W row stride 96, global mem)
__device__ __forceinline__ v16h fragA_W(const float* W, int obase, int cbase, int lane) {
  v16h a; int o = obase + (lane & 15);
#pragma unroll
  for (int i = 0; i < 16; ++i) a[i] = (_Float16)W[o * 96 + cbase + koffmap(lane, i)];
  return a;
}
// B[k=c][n=w] = S[c][w]   (stage matrix, f32 LDS stride LS)
__device__ __forceinline__ v16h fragB_S(const float* S, int nbase, int lane) {
  v16h v; int n = nbase + (lane & 15);
#pragma unroll
  for (int i = 0; i < 16; ++i) v[i] = (_Float16)S[koffmap(lane, i) * LS + n];
  return v;
}

// ---------------------------------------------------------------------------
__global__ void kzero(float* p, size_t n) {
  for (size_t i = blockIdx.x * (size_t)blockDim.x + threadIdx.x; i < n;
       i += (size_t)blockDim.x * gridDim.x)
    p[i] = 0.f;
}

// sci[b,d,n] = tanh(y)*sigmoid(y), y = sci_w @ X[...,0] + sci_b
__global__ void ksci(const float* __restrict__ X, const float* __restrict__ w,
                     const float* __restrict__ bi, float* __restrict__ sci) {
  int i = blockIdx.x * blockDim.x + threadIdx.x;
  if (i >= NB * NC * NN) return;
  int n = i % NN, d = (i / NN) % NC, b = i / (NN * NC);
  float y = bi[d];
  for (int c = 0; c < NC; ++c)
    y += w[d * NC + c] * X[(((size_t)b * NC + c) * NN + n) * NTIME];
  sci[i] = tanhf(y) * (1.f / (1.f + __expf(-y)));
}

// Dilated inception (k=2,3,6,7, crop to 58) + gating + sci factor.
// Writes Xct[b][t][c][n(pad 224)]  (pad pre-zeroed).
__global__ void kincept(const float* __restrict__ X, const float* __restrict__ sci,
                        const float* fw0, const float* fw1, const float* fw2, const float* fw3,
                        const float* fb0, const float* fb1, const float* fb2, const float* fb3,
                        const float* gw0, const float* gw1, const float* gw2, const float* gw3,
                        const float* gb0, const float* gb1, const float* gb2, const float* gb3,
                        float* __restrict__ Xct) {
  int i = blockIdx.x * blockDim.x + threadIdx.x;
  if (i >= NB * NC * NN * TOUT) return;
  int t = i % TOUT, n = (i / TOUT) % NN, oc = (i / (TOUT * NN)) % NC,
      b = i / (TOUT * NN * NC);
  const int KK[4] = {2, 3, 6, 7};
  int g = oc >> 3, j = oc & 7, k = KK[g];
  const float* fw = g == 0 ? fw0 : g == 1 ? fw1 : g == 2 ? fw2 : fw3;
  const float* gw = g == 0 ? gw0 : g == 1 ? gw1 : g == 2 ? gw2 : gw3;
  float f = (g == 0 ? fb0 : g == 1 ? fb1 : g == 2 ? fb2 : fb3)[j];
  float gg = (g == 0 ? gb0 : g == 1 ? gb1 : g == 2 ? gb2 : gb3)[j];
  int t0 = t + 7 - k;
  for (int c = 0; c < NC; ++c) {
    const float* xp = X + (((size_t)b * NC + c) * NN + n) * NTIME + t0;
    __builtin_prefetch(xp, 0, 0);
    const float* wf = fw + (j * NC + c) * k;
    const float* wg = gw + (j * NC + c) * k;
    for (int p = 0; p < k; ++p) { float xv = xp[p]; f += wf[p] * xv; gg += wg[p] * xv; }
  }
  float sc = sci[((size_t)b * NC + oc) * NN + n];
  Xct[(((size_t)b * TOUT + t) * NC + oc) * NPAD + n] =
      tanhf(f) * (1.f / (1.f + __expf(-gg))) * sc;
}

// x1/x2 (f16) for both mixprops: tanh(1x1 conv on Xc). Block per (b,t).
__global__ __launch_bounds__(256) void klin(
    const float* __restrict__ Xct,
    const float* w11, const float* b11, const float* w12, const float* b12,
    const float* w21, const float* b21, const float* w22, const float* b22,
    _Float16* x1a, _Float16* x2a, _Float16* x1b, _Float16* x2b) {
  __shared__ float tile[NC * LS];
  size_t base = (size_t)blockIdx.x * NC * NPAD;
  for (int i = threadIdx.x; i < NC * NPAD; i += 256) {
    int c = i / NPAD, v = i % NPAD;
    tile[c * LS + v] = Xct[base + i];
  }
  __syncthreads();
  for (int i = threadIdx.x; i < NC * NPAD; i += 256) {
    int c = i / NPAD, v = i % NPAD;
    float s11 = b11[c], s12 = b12[c], s21 = b21[c], s22 = b22[c];
    for (int k = 0; k < NC; ++k) {
      float xv = tile[k * LS + v];
      s11 += w11[c * NC + k] * xv; s12 += w12[c * NC + k] * xv;
      s21 += w21[c * NC + k] * xv; s22 += w22[c * NC + k] * xv;
    }
    bool ok = v < NN;
    x1a[base + i] = (_Float16)(ok ? tanhf(s11) : 0.f);
    x2a[base + i] = (_Float16)(ok ? tanhf(s12) : 0.f);
    x1b[base + i] = (_Float16)(ok ? tanhf(s21) : 0.f);
    x2b[base + i] = (_Float16)(ok ? tanhf(s22) : 0.f);
  }
}

// ---------------------------------------------------------------------------
// Fused Dy_MixProp per (t, b, mp): async-LDS staging -> adj (WMMA) -> softmax
// stats -> 2 dirs x GDEP=2 propagation (WMMA) -> fused 3-block MLP (WMMA) ->
// atomicAdd to Xg.  Dynamic LDS ~287KB.
// ---------------------------------------------------------------------------
__global__ __launch_bounds__(256) void kmix(
    const _Float16* __restrict__ x1a, const _Float16* __restrict__ x2a,
    const _Float16* __restrict__ x1b, const _Float16* __restrict__ x2b,
    const float* __restrict__ Xct,
    const float* m1w1, const float* m1b1, const float* m1w2, const float* m1b2,
    const float* m2w1, const float* m2b1, const float* m2w2, const float* m2b2,
    float* __restrict__ Xg) {
  extern __shared__ float sm[];
  float* E    = sm;                 // 224*228
  float* Xc   = E + NPAD * LS;      // 32*228
  float* HA   = Xc + NC * LS;       // 32*228
  float* HB   = HA + NC * LS;       // 32*228
  float* invr = HB + NC * LS;       // 224
  float* invc = invr + NPAD;        // 224
  float* redS = invc + NPAD;        // 16

  const int t = blockIdx.x, b = blockIdx.y, mp = blockIdx.z;
  const int tid = threadIdx.x, lane = tid & 31, wave = tid >> 5;
  const size_t bt = (size_t)b * TOUT + t;
  const _Float16* x1g = (mp ? x1b : x1a) + bt * NC * NPAD;
  const _Float16* x2g = (mp ? x2b : x2a) + bt * NC * NPAD;
  const float* Xcg = Xct + bt * NC * NPAD;
  _Float16* x1s = (_Float16*)HA;    // reuse HA/HB as f16 staging in phases A/B
  _Float16* x2s = (_Float16*)HB;

  // Phase A: async global->LDS staging (B128 chunks; rows are 16B-aligned:
  // f32 rows 896B -> LS*4=912B, f16 rows 448B -> XHS*2=464B).
  {
    const unsigned lXc = lds_addr32(Xc);
    const unsigned lX1 = lds_addr32(x1s);
    const unsigned lX2 = lds_addr32(x2s);
    for (int ch = tid; ch < NC * 56; ch += 256) {      // Xc: 56 x 16B per row
      int c = ch / 56, q = ch % 56;
      async_b128(lXc + c * (LS * 4) + q * 16, (const char*)Xcg + c * 896 + q * 16);
    }
    for (int ch = tid; ch < NC * 28; ch += 256) {      // x1/x2: 28 x 16B per row
      int c = ch / 28, q = ch % 28;
      async_b128(lX1 + c * (XHS * 2) + q * 16, (const char*)x1g + c * 448 + q * 16);
      async_b128(lX2 + c * (XHS * 2) + q * 16, (const char*)x2g + c * 448 + q * 16);
    }
    wait_async0();
  }
  __syncthreads();

  // Phase B: adj[v][w] = sum_c x1[c][v]*x2[c][w] via WMMA; track tile max
  float wmax = -3.4e38f;
  for (int tile = wave; tile < NTL * NTL; tile += 8) {
    int mt = tile / NTL, nt = tile % NTL;
    v16h va = fragA_xT(x1s, mt * 16, lane);
    v16h vb = fragB_x(x2s, nt * 16, lane);
    v8f d = {0.f, 0.f, 0.f, 0.f, 0.f, 0.f, 0.f, 0.f};
    d = wmma16(va, vb, d);
    int n = nt * 16 + (lane & 15);
#pragma unroll
    for (int r = 0; r < 8; ++r) {
      int m = mt * 16 + r + (lane < 16 ? 0 : 8);
      E[m * LS + n] = d[r];
      wmax = fmaxf(wmax, d[r]);
    }
  }
#pragma unroll
  for (int o = 16; o > 0; o >>= 1) wmax = fmaxf(wmax, __shfl_xor(wmax, o, 32));
  if (lane == 0) redS[wave] = wmax;
  __syncthreads();
  float M = redS[0];
#pragma unroll
  for (int w = 1; w < 8; ++w) M = fmaxf(M, redS[w]);

  // Phase C: E = exp(adj - M), zero padding; row/col softmax denominators
  for (int i = tid; i < NPAD * NPAD; i += 256) {
    int v = i / NPAD, w = i % NPAD;
    float val = (v < NN && w < NN) ? __expf(E[v * LS + w] - M) : 0.f;
    E[v * LS + w] = val;
  }
  __syncthreads();
  for (int v = tid; v < NPAD; v += 256) {
    float s = 0.f, s2 = 0.f;
    for (int w = 0; w < NN; ++w) s += E[v * LS + w];
    for (int u = 0; u < NN; ++u) s2 += E[u * LS + v];
    invr[v] = (v < NN) ? 1.f / s : 0.f;   // adj0: softmax over w (rows of E)
    invc[v] = (v < NN) ? 1.f / s2 : 0.f;  // adj1: uses E^T with col denominators
  }
  __syncthreads();

  const float* Wd[2] = {mp ? m2w1 : m1w1, mp ? m2w2 : m1w2};
  const float* Bd[2] = {mp ? m2b1 : m1b1, mp ? m2b2 : m1b2};

  for (int dir = 0; dir < 2; ++dir) {
    const float* inv = dir ? invc : invr;
    // GDEP step 1: HA = a*Xc + (1-a) * (Xc .* inv_k) @ (dir ? E^T : E)
    for (int tile = wave; tile < 2 * NTL; tile += 8) {
      int mt = tile / NTL, nt = tile % NTL;
      v8f d = {0.f, 0.f, 0.f, 0.f, 0.f, 0.f, 0.f, 0.f};
      for (int ks = 0; ks < 7; ++ks) {
        v16h va = fragA_hscaled(Xc, inv, mt * 16, ks * 32, lane);
        v16h vb = dir ? fragB_ET(E, ks * 32, nt * 16, lane)
                      : fragB_E(E, ks * 32, nt * 16, lane);
        d = wmma16(va, vb, d);
      }
      int n = nt * 16 + (lane & 15);
#pragma unroll
      for (int r = 0; r < 8; ++r) {
        int m = mt * 16 + r + (lane < 16 ? 0 : 8);
        HA[m * LS + n] = ALPHA_F * Xc[m * LS + n] + (1.f - ALPHA_F) * d[r];
      }
    }
    __syncthreads();
    // GDEP step 2: HB from HA
    for (int tile = wave; tile < 2 * NTL; tile += 8) {
      int mt = tile / NTL, nt = tile % NTL;
      v8f d = {0.f, 0.f, 0.f, 0.f, 0.f, 0.f, 0.f, 0.f};
      for (int ks = 0; ks < 7; ++ks) {
        v16h va = fragA_hscaled(HA, inv, mt * 16, ks * 32, lane);
        v16h vb = dir ? fragB_ET(E, ks * 32, nt * 16, lane)
                      : fragB_E(E, ks * 32, nt * 16, lane);
        d = wmma16(va, vb, d);
      }
      int n = nt * 16 + (lane & 15);
#pragma unroll
      for (int r = 0; r < 8; ++r) {
        int m = mt * 16 + r + (lane < 16 ? 0 : 8);
        HB[m * LS + n] = ALPHA_F * Xc[m * LS + n] + (1.f - ALPHA_F) * d[r];
      }
    }
    __syncthreads();
    // Fused MLP on concat[Xc;HA;HB] with 32x32 weight blocks, then atomicAdd
    const float* W = Wd[dir];
    const float* Bb = Bd[dir];
    for (int tile = wave; tile < 2 * NTL; tile += 8) {
      int mt = tile / NTL, nt = tile % NTL;
      v8f acc = {0.f, 0.f, 0.f, 0.f, 0.f, 0.f, 0.f, 0.f};
      acc = wmma16(fragA_W(W, mt * 16, 0, lane),  fragB_S(Xc, nt * 16, lane), acc);
      acc = wmma16(fragA_W(W, mt * 16, 32, lane), fragB_S(HA, nt * 16, lane), acc);
      acc = wmma16(fragA_W(W, mt * 16, 64, lane), fragB_S(HB, nt * 16, lane), acc);
      int n = nt * 16 + (lane & 15);
      if (n < NN) {
#pragma unroll
        for (int r = 0; r < 8; ++r) {
          int o = mt * 16 + r + (lane < 16 ? 0 : 8);
          atomicAdd(&Xg[(((size_t)b * NC + o) * NN + n) * TOUT + t], acc[r] + Bb[o]);
        }
      }
    }
    __syncthreads();
  }
}

// Skip conv: out2[b,o,n] = sum_{c,t} Xc * skip_w + skip_b + X_skip
__global__ void kskip(const float* __restrict__ Xct, const float* __restrict__ sw,
                      const float* __restrict__ sb, const float* __restrict__ Xsk,
                      float* __restrict__ out2) {
  __shared__ float tile[TOUT * NC];
  int b = blockIdx.x / NN, n = blockIdx.x % NN;
  for (int i = threadIdx.x; i < TOUT * NC; i += 64) {
    int t = i / NC, c = i % NC;
    tile[i] = Xct[(((size_t)b * TOUT + t) * NC + c) * NPAD + n];
  }
  __syncthreads();
  int o = threadIdx.x;
  float s = sb[o];
  for (int t = 0; t < TOUT; ++t)
    for (int c = 0; c < NC; ++c)
      s += tile[t * NC + c] * sw[((size_t)o * NC + c) * TOUT + t];
  size_t oi = ((size_t)b * 64 + o) * NN + n;
  out2[oi] = s + Xsk[oi];
}

// LayerNorm pass 1: per-batch sum / sumsq of (Xg + residual)
__global__ void kred(const float* __restrict__ Xg, const float* __restrict__ X,
                     float* __restrict__ red) {
  const int PB = NC * NN * TOUT;  // 371200
  int b = blockIdx.y;
  float s = 0.f, s2 = 0.f;
  for (int i = blockIdx.x * blockDim.x + threadIdx.x; i < PB;
       i += blockDim.x * gridDim.x) {
    int t = i % TOUT, n = (i / TOUT) % NN, c = i / (TOUT * NN);
    float v = Xg[(size_t)b * PB + i] +
              X[(((size_t)b * NC + c) * NN + n) * NTIME + 6 + t];
    s += v; s2 += v * v;
  }
  __shared__ float sh[512];
  sh[threadIdx.x] = s; sh[256 + threadIdx.x] = s2;
  __syncthreads();
  for (int o = 128; o > 0; o >>= 1) {
    if (threadIdx.x < o) {
      sh[threadIdx.x] += sh[threadIdx.x + o];
      sh[256 + threadIdx.x] += sh[256 + threadIdx.x + o];
    }
    __syncthreads();
  }
  if (threadIdx.x == 0) { atomicAdd(&red[2 * b], sh[0]); atomicAdd(&red[2 * b + 1], sh[256]); }
}

// LayerNorm pass 2: normalize + node-indexed affine -> out1
__global__ void kln(const float* __restrict__ Xg, const float* __restrict__ X,
                    const float* __restrict__ red, const int* __restrict__ idx,
                    const float* __restrict__ lnw, const float* __restrict__ lnb,
                    float* __restrict__ out1) {
  const int PB = NC * NN * TOUT;
  const float invn = 1.f / (float)PB;
  for (size_t i = blockIdx.x * (size_t)blockDim.x + threadIdx.x;
       i < (size_t)NB * PB; i += (size_t)blockDim.x * gridDim.x) {
    int b = (int)(i / PB), r = (int)(i % PB);
    int t = r % TOUT, n = (r / TOUT) % NN, c = r / (TOUT * NN);
    float pre = Xg[i] + X[(((size_t)b * NC + c) * NN + n) * NTIME + 6 + t];
    float mu = red[2 * b] * invn;
    float var = red[2 * b + 1] * invn - mu * mu;
    size_t wix = ((size_t)c * NN + idx[n]) * TOUT + t;
    out1[i] = (pre - mu) * rsqrtf(var + 1e-5f) * lnw[wix] + lnb[wix];
  }
}

// ---------------------------------------------------------------------------
extern "C" void kernel_launch(void* const* d_in, const int* in_sizes, int n_in,
                              void* d_out, int out_size, void* d_ws, size_t ws_size,
                              hipStream_t stream) {
  const float* X    = (const float*)d_in[0];
  const float* Xsk  = (const float*)d_in[1];
  const int*   idx  = (const int*)d_in[3];
  const float* sciw = (const float*)d_in[5];
  const float* scib = (const float*)d_in[6];
  const float *fw0 = (const float*)d_in[7],  *fw1 = (const float*)d_in[8],
              *fw2 = (const float*)d_in[9],  *fw3 = (const float*)d_in[10];
  const float *fb0 = (const float*)d_in[11], *fb1 = (const float*)d_in[12],
              *fb2 = (const float*)d_in[13], *fb3 = (const float*)d_in[14];
  const float *gw0 = (const float*)d_in[15], *gw1 = (const float*)d_in[16],
              *gw2 = (const float*)d_in[17], *gw3 = (const float*)d_in[18];
  const float *gb0 = (const float*)d_in[19], *gb1 = (const float*)d_in[20],
              *gb2 = (const float*)d_in[21], *gb3 = (const float*)d_in[22];
  const float* skw = (const float*)d_in[23];
  const float* skb = (const float*)d_in[24];
  const float *m1l1w = (const float*)d_in[25], *m1l1b = (const float*)d_in[26],
              *m1l2w = (const float*)d_in[27], *m1l2b = (const float*)d_in[28],
              *m1m1w = (const float*)d_in[29], *m1m1b = (const float*)d_in[30],
              *m1m2w = (const float*)d_in[31], *m1m2b = (const float*)d_in[32];
  const float *m2l1w = (const float*)d_in[33], *m2l1b = (const float*)d_in[34],
              *m2l2w = (const float*)d_in[35], *m2l2b = (const float*)d_in[36],
              *m2m1w = (const float*)d_in[37], *m2m1b = (const float*)d_in[38],
              *m2m2w = (const float*)d_in[39], *m2m2b = (const float*)d_in[40];
  const float* lnw = (const float*)d_in[41];
  const float* lnb = (const float*)d_in[42];

  float* out1 = (float*)d_out;                               // Xo
  float* out2 = out1 + (size_t)NB * NC * NN * TOUT;          // X_skip_out

  // workspace layout (bytes)
  char* ws = (char*)d_ws;
  const size_t XCT_N = (size_t)NB * TOUT * NC * NPAD;        // 6,651,904 f32
  float*    Xct = (float*)ws;
  _Float16* x1a = (_Float16*)(ws + 26607616);
  _Float16* x2a = (_Float16*)(ws + 39911424);
  _Float16* x1b = (_Float16*)(ws + 53215232);
  _Float16* x2b = (_Float16*)(ws + 66519040);
  float*    Xg  = (float*)(ws + 79822848);                   // 5,939,200 f32
  float*    sci = (float*)(ws + 103579648);                  // 102,400 f32
  float*    red = (float*)(ws + 103989248);                  // 32 f32

  const size_t XG_N = (size_t)NB * NC * NN * TOUT;

  kzero<<<2048, 256, 0, stream>>>(Xct, XCT_N);
  kzero<<<2048, 256, 0, stream>>>(Xg, XG_N);
  kzero<<<1, 32, 0, stream>>>(red, 32);

  ksci<<<(NB * NC * NN + 255) / 256, 256, 0, stream>>>(X, sciw, scib, sci);

  int nout = NB * NC * NN * TOUT;
  kincept<<<(nout + 255) / 256, 256, 0, stream>>>(
      X, sci, fw0, fw1, fw2, fw3, fb0, fb1, fb2, fb3,
      gw0, gw1, gw2, gw3, gb0, gb1, gb2, gb3, Xct);

  klin<<<NB * TOUT, 256, 0, stream>>>(Xct, m1l1w, m1l1b, m1l2w, m1l2b,
                                      m2l1w, m2l1b, m2l2w, m2l2b,
                                      x1a, x2a, x1b, x2b);

  const size_t MIX_LDS =
      (size_t)(NPAD * LS + 3 * NC * LS + 2 * NPAD + 16) * sizeof(float);  // ~287KB
  kmix<<<dim3(TOUT, NB, 2), 256, MIX_LDS, stream>>>(
      x1a, x2a, x1b, x2b, Xct,
      m1m1w, m1m1b, m1m2w, m1m2b, m2m1w, m2m1b, m2m2w, m2m2b, Xg);

  kskip<<<NB * NN, 64, 0, stream>>>(Xct, skw, skb, Xsk, out2);

  kred<<<dim3(64, NB), 256, 0, stream>>>(Xg, X, red);
  kln<<<(int)((XG_N + 255) / 256), 256, 0, stream>>>(Xg, X, red, idx, lnw, lnb, out1);

  (void)in_sizes; (void)n_in; (void)out_size; (void)ws_size;
}
